// GraphGRU_35158602285575
// MI455X (gfx1250) — matrix operand order repaired and checked by reading, not compiled, not run on hardware
//
#include <hip/hip_runtime.h>
#include <hip/hip_bf16.h>

#define M_MSG 30000
#define H_DIM 512
#define K_NEI 8
#define DEPTH_N 8
#define LDS_STRIDE 520   // 512 + 8 bf16 pad -> 1040B row stride, conflict-free b128 ds reads

typedef __attribute__((ext_vector_type(16))) __bf16 v16bf;
typedef __attribute__((ext_vector_type(8)))  __bf16 bf16x8;
typedef __attribute__((ext_vector_type(4)))  __bf16 bf16x4;
typedef __attribute__((ext_vector_type(8)))  float  v8f;

__device__ __forceinline__ float sigmoidf(float v) {
    return 1.0f / (1.0f + __expf(-v));
}
__device__ __forceinline__ float tanh_fast(float v) {
    float e = __expf(-2.0f * fabsf(v));
    float t = (1.0f - e) / (1.0f + e);
    return copysignf(t, v);
}

// ---- WMMA fragment helpers (CDNA5 16x16x32 bf16 layouts, wave32) ----

// A (16x32, bf16) from LDS row-major [16][LDS_STRIDE]:
// lanes 0-15 : row = lane,     elems 0-15 -> K = k0+0..15
// lanes 16-31: row = lane-16,  elems 0-15 -> K = k0+16..31
__device__ __forceinline__ v16bf load_a_lds(const __bf16* lds, int lane, int k0) {
    int r   = lane & 15;
    int hi  = lane >> 4;
    int kb0 = k0 + hi * 16;
    int kb1 = kb0 + 8;
    bf16x8 lo = *(const bf16x8*)(lds + r * LDS_STRIDE + kb0);
    bf16x8 hv = *(const bf16x8*)(lds + r * LDS_STRIDE + kb1);
    v16bf a;
#pragma unroll
    for (int i = 0; i < 8; ++i) { a[i] = lo[i]; a[8 + i] = hv[i]; }
    return a;
}

// B (32x16, bf16) with B[k][n] = W[n0+n][k0+k], W row-major [512][512] bf16:
// lanes 0-15 hold K = k0+0..15, lanes 16-31 hold K = k0+16..31 -> one 32B contiguous load.
__device__ __forceinline__ v16bf load_b_w(const __bf16* W, int lane, int n0, int k0) {
    int col = n0 + (lane & 15);
    int kb  = k0 + ((lane >> 4) << 4);
    return *(const v16bf*)(W + (size_t)col * H_DIM + kb);
}

__device__ __forceinline__ v8f wmma_bf16(v16bf a, v16bf b, v8f c) {
    return __builtin_amdgcn_wmma_f32_16x16x32_bf16(false, a, false, b, (short)0, c, false, false);
}

// ---- Weight repack: fp32 (row stride rs, col offset co) -> bf16 row-major [512][512] ----
__global__ __launch_bounds__(256) void cvt_w_kernel(const float* __restrict__ src, int rs, int co,
                                                    __bf16* __restrict__ dst) {
    int i = blockIdx.x * 256 + threadIdx.x;          // 0 .. 512*512-1
    int r = i >> 9, c = i & 511;
    dst[i] = (__bf16)src[(size_t)r * rs + co + c];
}

// ---- C[M,512] = A[M,512] @ W^T (+bias), bf16 WMMA, fp32 accumulate ----
__global__ __launch_bounds__(128) void gemm_bias_kernel(const float* __restrict__ A,
                                                        const __bf16* __restrict__ W,
                                                        const float* __restrict__ bias,
                                                        float* __restrict__ C) {
    __shared__ __bf16 As[16 * LDS_STRIDE];
    const int m0 = blockIdx.x * 16;
    const int t  = threadIdx.x;

    // Stage A tile (16x512 fp32 -> bf16 LDS); each thread: 4 cols per row.
#pragma unroll 4
    for (int r = 0; r < 16; ++r) {
        float4 v = *(const float4*)(A + (size_t)(m0 + r) * H_DIM + t * 4);
        bf16x4 bv = { (__bf16)v.x, (__bf16)v.y, (__bf16)v.z, (__bf16)v.w };
        *(bf16x4*)(&As[r * LDS_STRIDE + t * 4]) = bv;
    }
    __syncthreads();

    const int wave = t >> 5, lane = t & 31;
    const int lrow = lane & 15, lhi = lane >> 4;

    for (int nt = 0; nt < 8; ++nt) {
        const int n0 = wave * 128 + nt * 16;
        v8f acc = {0.f, 0.f, 0.f, 0.f, 0.f, 0.f, 0.f, 0.f};
#pragma unroll 4
        for (int k0 = 0; k0 < H_DIM; k0 += 32) {
            v16bf a = load_a_lds(As, lane, k0);
            v16bf b = load_b_w(W, lane, n0, k0);
            acc = wmma_bf16(a, b, acc);
        }
        const int col   = n0 + lrow;
        const int rbase = m0 + lhi * 8;
        const float bv  = bias ? bias[col] : 0.0f;
#pragma unroll
        for (int e = 0; e < 8; ++e) {
            C[(size_t)(rbase + e) * H_DIM + col] = acc[e] + bv;
        }
    }
}

// ---- Gather/reduce: sum_h[m] = sum_k h[g[m,k]]; sum_g[m] = sum_k sigmoid(r1b[m]+UH[g[m,k]])*h[g[m,k]] ----
__global__ __launch_bounds__(256) void gather_kernel(const float* __restrict__ h,
                                                     const float* __restrict__ uh,
                                                     const float* __restrict__ r1b,
                                                     const int* __restrict__ g,
                                                     float* __restrict__ sum_h,
                                                     float* __restrict__ sum_g) {
    const int m = blockIdx.x;
    const int c = threadIdx.x * 2;
    const size_t base = (size_t)m * H_DIM + c;
    float2 r1v = *(const float2*)(r1b + base);
    float shx = 0.f, shy = 0.f, sgx = 0.f, sgy = 0.f;
    int idx[K_NEI];
#pragma unroll
    for (int k = 0; k < K_NEI; ++k) idx[k] = g[m * K_NEI + k];
#pragma unroll
    for (int k = 0; k < K_NEI; ++k) {
        const size_t jb = (size_t)idx[k] * H_DIM + c;
        float2 hv = *(const float2*)(h + jb);
        float2 uv = *(const float2*)(uh + jb);
        float rx = sigmoidf(r1v.x + uv.x);
        float ry = sigmoidf(r1v.y + uv.y);
        shx += hv.x;      shy += hv.y;
        sgx += rx * hv.x; sgy += ry * hv.y;
    }
    float2 o1 = {shx, shy}, o2 = {sgx, sgy};
    *(float2*)(sum_h + base) = o1;
    *(float2*)(sum_g + base) = o2;
}

// ---- Fused gate update: z = sigmoid(zx + sum_h@Wzh^T); pre = tanh(hx + sum_g@Whh^T);
//      h_out = ((1-z)*sum_h + z*pre) * mask(row!=0) ----
__global__ __launch_bounds__(128) void gate_fused_kernel(const float* __restrict__ sum_h,
                                                         const float* __restrict__ sum_g,
                                                         const __bf16* __restrict__ Wzh,
                                                         const __bf16* __restrict__ Whh,
                                                         const float* __restrict__ zx,
                                                         const float* __restrict__ hx,
                                                         float* __restrict__ h_out) {
    __shared__ __bf16 As[16 * LDS_STRIDE];   // sum_h tile
    __shared__ __bf16 Bs[16 * LDS_STRIDE];   // sum_g tile
    const int m0 = blockIdx.x * 16;
    const int t  = threadIdx.x;

#pragma unroll 4
    for (int r = 0; r < 16; ++r) {
        const size_t gi = (size_t)(m0 + r) * H_DIM + t * 4;
        float4 a = *(const float4*)(sum_h + gi);
        float4 b = *(const float4*)(sum_g + gi);
        bf16x4 av = { (__bf16)a.x, (__bf16)a.y, (__bf16)a.z, (__bf16)a.w };
        bf16x4 bv = { (__bf16)b.x, (__bf16)b.y, (__bf16)b.z, (__bf16)b.w };
        *(bf16x4*)(&As[r * LDS_STRIDE + t * 4]) = av;
        *(bf16x4*)(&Bs[r * LDS_STRIDE + t * 4]) = bv;
    }
    __syncthreads();

    const int wave = t >> 5, lane = t & 31;
    const int lrow = lane & 15, lhi = lane >> 4;

    for (int nt = 0; nt < 8; ++nt) {
        const int n0 = wave * 128 + nt * 16;
        v8f accz = {0.f, 0.f, 0.f, 0.f, 0.f, 0.f, 0.f, 0.f};
        v8f accp = {0.f, 0.f, 0.f, 0.f, 0.f, 0.f, 0.f, 0.f};
#pragma unroll 2
        for (int k0 = 0; k0 < H_DIM; k0 += 32) {
            v16bf a1 = load_a_lds(As, lane, k0);
            v16bf b1 = load_b_w(Wzh, lane, n0, k0);
            accz = wmma_bf16(a1, b1, accz);
            v16bf a2 = load_a_lds(Bs, lane, k0);
            v16bf b2 = load_b_w(Whh, lane, n0, k0);
            accp = wmma_bf16(a2, b2, accp);
        }
        const int col   = n0 + lrow;
        const int rbase = m0 + lhi * 8;
#pragma unroll
        for (int e = 0; e < 8; ++e) {
            const int m = rbase + e;
            const size_t ix = (size_t)m * H_DIM + col;
            float z  = sigmoidf(zx[ix] + accz[e]);
            float p  = tanh_fast(hx[ix] + accp[e]);
            float sh = sum_h[ix];
            float hv = (1.0f - z) * sh + z * p;
            if (m == 0) hv = 0.0f;   // mask row 0 (null message)
            h_out[ix] = hv;
        }
    }
}

extern "C" void kernel_launch(void* const* d_in, const int* in_sizes, int n_in,
                              void* d_out, int out_size, void* d_ws, size_t ws_size,
                              hipStream_t stream) {
    const float* h_in  = (const float*)d_in[0];
    const float* x     = (const float*)d_in[1];
    const int*   graph = (const int*)d_in[2];
    const float* Wz_w  = (const float*)d_in[3];   // [512, 1024]
    const float* Wz_b  = (const float*)d_in[4];
    const float* Wr_w  = (const float*)d_in[5];   // [512, 512]
    const float* Ur_w  = (const float*)d_in[6];   // [512, 512]
    const float* Ur_b  = (const float*)d_in[7];
    const float* Wh_w  = (const float*)d_in[8];   // [512, 1024]
    const float* Wh_b  = (const float*)d_in[9];
    (void)in_sizes; (void)n_in; (void)out_size; (void)ws_size;

    char* ws = (char*)d_ws;
    size_t off = 0;
    auto bump = [&](size_t bytes) -> void* {
        void* p = ws + off;
        off = (off + bytes + 255) & ~(size_t)255;
        return p;
    };
    const size_t WB  = (size_t)H_DIM * H_DIM * sizeof(__bf16);
    const size_t MHB = (size_t)M_MSG * H_DIM * sizeof(float);

    __bf16* Wr_bf  = (__bf16*)bump(WB);
    __bf16* Ur_bf  = (__bf16*)bump(WB);
    __bf16* Wzx_bf = (__bf16*)bump(WB);
    __bf16* Wzh_bf = (__bf16*)bump(WB);
    __bf16* Whx_bf = (__bf16*)bump(WB);
    __bf16* Whh_bf = (__bf16*)bump(WB);
    float* r1b   = (float*)bump(MHB);
    float* zx    = (float*)bump(MHB);
    float* hx    = (float*)bump(MHB);
    float* uh    = (float*)bump(MHB);
    float* sum_h = (float*)bump(MHB);
    float* sum_g = (float*)bump(MHB);
    float* hbuf0 = (float*)bump(MHB);
    float* hbuf1 = (float*)bump(MHB);

    const dim3 cvtGrid((H_DIM * H_DIM) / 256), cvtBlk(256);
    const dim3 gemmGrid(M_MSG / 16), gemmBlk(128);
    const dim3 gatGrid(M_MSG), gatBlk(256);

    // Weight repack to bf16 (once).
    cvt_w_kernel<<<cvtGrid, cvtBlk, 0, stream>>>(Wr_w, H_DIM,     0,     Wr_bf);
    cvt_w_kernel<<<cvtGrid, cvtBlk, 0, stream>>>(Ur_w, H_DIM,     0,     Ur_bf);
    cvt_w_kernel<<<cvtGrid, cvtBlk, 0, stream>>>(Wz_w, 2 * H_DIM, 0,     Wzx_bf);
    cvt_w_kernel<<<cvtGrid, cvtBlk, 0, stream>>>(Wz_w, 2 * H_DIM, H_DIM, Wzh_bf);
    cvt_w_kernel<<<cvtGrid, cvtBlk, 0, stream>>>(Wh_w, 2 * H_DIM, 0,     Whx_bf);
    cvt_w_kernel<<<cvtGrid, cvtBlk, 0, stream>>>(Wh_w, 2 * H_DIM, H_DIM, Whh_bf);

    // Loop-invariant x-projections (hoisted out of the depth loop).
    gemm_bias_kernel<<<gemmGrid, gemmBlk, 0, stream>>>(x, Wr_bf,  Ur_b, r1b);
    gemm_bias_kernel<<<gemmGrid, gemmBlk, 0, stream>>>(x, Wzx_bf, Wz_b, zx);
    gemm_bias_kernel<<<gemmGrid, gemmBlk, 0, stream>>>(x, Whx_bf, Wh_b, hx);

    const float* cur = h_in;
    for (int it = 0; it < DEPTH_N; ++it) {
        // UH = h @ Ur^T once per step; r2[m,k] = UH[graph[m,k]]  (8x FLOP cut vs reference)
        gemm_bias_kernel<<<gemmGrid, gemmBlk, 0, stream>>>(cur, Ur_bf, nullptr, uh);
        gather_kernel<<<gatGrid, gatBlk, 0, stream>>>(cur, uh, r1b, graph, sum_h, sum_g);
        float* nxt = (it == DEPTH_N - 1) ? (float*)d_out : ((it & 1) ? hbuf1 : hbuf0);
        gate_fused_kernel<<<gemmGrid, gemmBlk, 0, stream>>>(sum_h, sum_g, Wzh_bf, Whh_bf,
                                                            zx, hx, nxt);
        cur = nxt;
    }
}
